// LovaszSoftmaxLoss_274877907422
// MI455X (gfx1250) — matrix operand order repaired and checked by reading, not compiled, not run on hardware
//
#include <hip/hip_runtime.h>
#include <cstdint>

typedef __attribute__((ext_vector_type(16))) _Float16 v16h;
typedef __attribute__((ext_vector_type(8)))  float    v8f;

// Problem constants: B=8, C=8, H=W=512  ->  P = 8*512*512 = 2^21 pixels.
#define PPIX (1 << 21)
#define HWSZ (1 << 18)
#define NBLK 512            /* 2^21 / 4096 chunks per class */
#define NCLS 7              /* classes 1..7 (class 0 = ignore, never "present") */

// -------------------------------------------------------------------------
// Shared helpers
// -------------------------------------------------------------------------
__device__ __forceinline__ unsigned block_scan_incl_u32(unsigned v, unsigned* buf) {
    int t = threadIdx.x;
    buf[t] = v;
    __syncthreads();
#pragma unroll
    for (int off = 1; off < 256; off <<= 1) {
        unsigned add = (t >= off) ? buf[t - off] : 0u;
        __syncthreads();
        buf[t] += add;
        __syncthreads();
    }
    unsigned r = buf[t];
    __syncthreads();
    return r;
}

// -------------------------------------------------------------------------
// Phase 1: softmax + sortable key construction.
// key = ~((float_bits(err) << 1) | fg)  -> ascending sort == descending err.
// -------------------------------------------------------------------------
__global__ void build_keys(const float* __restrict__ lg, const int* __restrict__ tg,
                           unsigned* __restrict__ keys, int cbase, int gg) {
    int p = blockIdx.x * 256 + threadIdx.x;
    int b = p >> 18;
    int q = p & (HWSZ - 1);
    const float* base = lg + (((size_t)b) << 21) + q;
    float l[8];
#pragma unroll
    for (int c = 0; c < 8; c++) l[c] = base[(size_t)c << 18];
    float m = l[0];
#pragma unroll
    for (int c = 1; c < 8; c++) m = fmaxf(m, l[c]);
    float e[8];
    float s = 0.f;
#pragma unroll
    for (int c = 0; c < 8; c++) { e[c] = __expf(l[c] - m); s += e[c]; }
    float inv = 1.f / s;
    int t = tg[p];
    bool valid = (t != 0);
    for (int sidx = 0; sidx < gg; sidx++) {
        int c = cbase + sidx + 1;               // logical class 1..7
        float prob = e[c] * inv;
        unsigned fg = (valid && (t == c)) ? 1u : 0u;
        float err = valid ? (fg ? (1.f - prob) : prob) : 0.f;
        unsigned kb = (__float_as_uint(err) << 1) | fg;
        keys[(size_t)sidx * PPIX + p] = ~kb;
    }
}

// -------------------------------------------------------------------------
// Radix sort (4-bit LSD, 8 passes, stable). Chunk = 4096 elems per block.
// -------------------------------------------------------------------------
__global__ void radix_hist(const unsigned* __restrict__ src, unsigned* __restrict__ hist,
                           int pass) {
    __shared__ unsigned cnt[16];
    int t = threadIdx.x, slot = blockIdx.y, blk = blockIdx.x;
    if (t < 16) cnt[t] = 0u;
    __syncthreads();
    const unsigned* p = src + (size_t)slot * PPIX + (size_t)blk * 4096;
    unsigned sh = pass * 4;
#pragma unroll
    for (int j = 0; j < 16; j++) {
        unsigned k = p[j * 256 + t];
        atomicAdd(&cnt[(k >> sh) & 15u], 1u);
    }
    __syncthreads();
    if (t < 16) hist[(size_t)slot * (16 * NBLK) + (size_t)t * NBLK + blk] = cnt[t];
}

__global__ void scan_hist(const unsigned* __restrict__ hist, unsigned* __restrict__ base) {
    __shared__ unsigned sbuf[256];
    __shared__ unsigned carry;
    int t = threadIdx.x, slot = blockIdx.x;
    if (t == 0) carry = 0u;
    __syncthreads();
    const unsigned* h = hist + (size_t)slot * (16 * NBLK);
    unsigned* bs = base + (size_t)slot * (16 * NBLK);
    for (int tile = 0; tile < 32; tile++) {
        unsigned v = h[tile * 256 + t];
        unsigned incl = block_scan_incl_u32(v, sbuf);
        bs[tile * 256 + t] = incl - v + carry;
        __syncthreads();
        if (t == 255) carry += incl;
        __syncthreads();
    }
}

__global__ void radix_scatter(const unsigned* __restrict__ src, unsigned* __restrict__ dst,
                              const unsigned* __restrict__ base, int pass) {
    __shared__ unsigned sbuf[256];
    int t = threadIdx.x, slot = blockIdx.y, blk = blockIdx.x;
    unsigned sh = pass * 4;
    const uint4* s4 = (const uint4*)(src + (size_t)slot * PPIX + (size_t)blk * 4096);
    unsigned k[16];
#pragma unroll
    for (int jj = 0; jj < 4; jj++) {
        uint4 q = s4[t * 4 + jj];
        k[jj * 4 + 0] = q.x; k[jj * 4 + 1] = q.y;
        k[jj * 4 + 2] = q.z; k[jj * 4 + 3] = q.w;
    }
    // Per-thread stable ranks via 16 byte-counters packed in two u64.
    unsigned long long lo = 0ull, hi = 0ull;
    unsigned rank[16];
#pragma unroll
    for (int j = 0; j < 16; j++) {
        unsigned d = (k[j] >> sh) & 15u;
        unsigned bsh = (d & 7u) * 8u;
        if (d < 8u) { rank[j] = (unsigned)((lo >> bsh) & 0xFFull); lo += (1ull << bsh); }
        else        { rank[j] = (unsigned)((hi >> bsh) & 0xFFull); hi += (1ull << bsh); }
    }
    // Cross-thread exclusive scans: pack 2 digits per word as 16-bit lanes.
    unsigned exclP[8];
#pragma unroll
    for (int w = 0; w < 8; w++) {
        unsigned d0 = 2 * w, d1 = 2 * w + 1;
        unsigned c0 = (d0 < 8) ? (unsigned)((lo >> (d0 * 8)) & 0xFFull)
                               : (unsigned)((hi >> ((d0 - 8) * 8)) & 0xFFull);
        unsigned c1 = (d1 < 8) ? (unsigned)((lo >> (d1 * 8)) & 0xFFull)
                               : (unsigned)((hi >> ((d1 - 8) * 8)) & 0xFFull);
        unsigned packv = c0 | (c1 << 16);
        unsigned incl = block_scan_incl_u32(packv, sbuf);
        exclP[w] = incl - packv;
    }
    const unsigned* bb = base + (size_t)slot * (16 * NBLK) + blk;
    unsigned* dd = dst + (size_t)slot * PPIX;
#pragma unroll
    for (int j = 0; j < 16; j++) {
        unsigned d = (k[j] >> sh) & 15u;
        unsigned te = (exclP[d >> 1] >> ((d & 1u) * 16u)) & 0xFFFFu;
        unsigned idx = bb[d * NBLK] + te + rank[j];
        dd[idx] = k[j];
    }
}

// -------------------------------------------------------------------------
// Phase 3: fg prefix counts over sorted order.
// -------------------------------------------------------------------------
__global__ void fg_count(const unsigned* __restrict__ keys, unsigned* __restrict__ blockFg) {
    __shared__ unsigned sbuf[256];
    int t = threadIdx.x, slot = blockIdx.y, blk = blockIdx.x;
    const unsigned* p = keys + (size_t)slot * PPIX + (size_t)blk * 4096;
    unsigned s = 0;
#pragma unroll
    for (int j = 0; j < 16; j++) s += ((~p[j * 256 + t]) & 1u);
    sbuf[t] = s;
    __syncthreads();
#pragma unroll
    for (int off = 128; off > 0; off >>= 1) {
        if (t < off) sbuf[t] += sbuf[t + off];
        __syncthreads();
    }
    if (t == 0) blockFg[(size_t)slot * NBLK + blk] = sbuf[0];
}

__global__ void scan_fg(const unsigned* __restrict__ blockFg, unsigned* __restrict__ blockFgBase,
                        unsigned* __restrict__ gts, int cbase) {
    __shared__ unsigned sbuf[256];
    __shared__ unsigned carry;
    int t = threadIdx.x, slot = blockIdx.x;
    if (t == 0) carry = 0u;
    __syncthreads();
    for (int tile = 0; tile < 2; tile++) {
        unsigned v = blockFg[(size_t)slot * NBLK + tile * 256 + t];
        unsigned incl = block_scan_incl_u32(v, sbuf);
        blockFgBase[(size_t)slot * NBLK + tile * 256 + t] = incl - v + carry;
        __syncthreads();
        if (t == 255) carry += incl;
        __syncthreads();
    }
    if (t == 0) gts[cbase + slot] = carry;
}

// -------------------------------------------------------------------------
// Phase 4: Lovasz accumulation. Fine-grained 16-wide inclusive scans on the
// matrix unit: D = [L|0](16x32,f16) x [X;0](32x16,f16), one v_wmma per tile
// of 256 elements (2 tiles per wave, 16 per block).
// -------------------------------------------------------------------------
__global__ void lovasz_block(const unsigned* __restrict__ keys,
                             const unsigned* __restrict__ blockFgBase,
                             const unsigned* __restrict__ gtsArr,
                             float* __restrict__ blockLoss, int cbase) {
    __shared__ _Float16 fgLds[4096];
    __shared__ float errLds[4096];
    __shared__ float segLds[256];
    __shared__ float sbf[256];
    int t = threadIdx.x, slot = blockIdx.y, blk = blockIdx.x;
    const unsigned* p = keys + (size_t)slot * PPIX + (size_t)blk * 4096;
#pragma unroll
    for (int j = 0; j < 16; j++) {
        unsigned kb = ~p[j * 256 + t];
        int e = j * 256 + t;
        fgLds[e] = (_Float16)(float)(kb & 1u);
        errLds[e] = __uint_as_float(kb >> 1);
    }
    __syncthreads();

    int ln = t & 31;
    int wv = t >> 5;
    int m = ln & 15;
    int kb0 = (ln < 16) ? 0 : 8;
    // A = [L | 0]: row m, lanes 0..15 halves j=0..15 carry K=j, lanes 16..31 zero.
    v16h a;
#pragma unroll
    for (int j = 0; j < 16; j++)
        a[j] = (ln < 16 && j <= m) ? (_Float16)1.0f : (_Float16)0.0f;

    v8f dAcc[2];
#pragma unroll
    for (int tt = 0; tt < 2; tt++) {
        int T = wv * 2 + tt;
        // B = [X ; 0]: lanes 0..15 = column n, halves = rows K=0..15 of X.
        v16h b;
        if (ln < 16) {
            b = *(const v16h*)&fgLds[T * 256 + ln * 16];
        } else {
#pragma unroll
            for (int j = 0; j < 16; j++) b[j] = (_Float16)0.0f;
        }
        v8f c = {};
        dAcc[tt] = __builtin_amdgcn_wmma_f32_16x16x32_f16(
            false, a, false, b, (short)0, c, false, false);
        // Row 15 of D = segment totals -> lanes 16..31, element 7.
        if (ln >= 16) segLds[T * 16 + (ln - 16)] = dAcc[tt][7];
    }
    __syncthreads();

    // Exclusive scan of the 256 segment sums.
    {
        float v = segLds[t];
        sbf[t] = v;
        __syncthreads();
#pragma unroll
        for (int off = 1; off < 256; off <<= 1) {
            float add = (t >= off) ? sbf[t - off] : 0.f;
            __syncthreads();
            sbf[t] += add;
            __syncthreads();
        }
        float excl = sbf[t] - v;
        __syncthreads();
        segLds[t] = excl;
    }
    __syncthreads();

    float bb = (float)blockFgBase[(size_t)slot * NBLK + blk];
    float gts = (float)gtsArr[cbase + slot];
    float acc = 0.f;
#pragma unroll
    for (int tt = 0; tt < 2; tt++) {
        int T = wv * 2 + tt;
        int col = ln & 15;
        float segBase = segLds[T * 16 + col];
#pragma unroll
        for (int v = 0; v < 8; v++) {
            int row = (ln < 16) ? v : (8 + v);
            int e = T * 256 + col * 16 + row;
            float err = errLds[e];
            if (err > 0.f) {
                float Fi = bb + segBase + dAcc[tt][v];   // inclusive fg prefix
                float fgv = (float)fgLds[e];
                int gi = blk * 4096 + e;                 // global sorted position
                float ip1 = (float)(gi + 1);
                float jac = 1.f - (gts - Fi) / (gts + ip1 - Fi);
                float g;
                if (gi == 0) {
                    g = jac;
                } else {
                    float Fim1 = Fi - fgv;
                    float jacp = 1.f - (gts - Fim1) / (gts + (ip1 - 1.f) - Fim1);
                    g = jac - jacp;
                }
                acc += err * g;
            }
        }
    }
    // Deterministic block reduction.
    sbf[t] = acc;
    __syncthreads();
#pragma unroll
    for (int off = 128; off > 0; off >>= 1) {
        if (t < off) sbf[t] += sbf[t + off];
        __syncthreads();
    }
    if (t == 0) blockLoss[(size_t)slot * NBLK + blk] = sbf[0];
}

__global__ void reduce_loss(const float* __restrict__ blockLoss, float* __restrict__ lossArr,
                            int cbase) {
    __shared__ float sbf[256];
    int t = threadIdx.x, slot = blockIdx.x;
    float v = blockLoss[(size_t)slot * NBLK + t] + blockLoss[(size_t)slot * NBLK + 256 + t];
    sbf[t] = v;
    __syncthreads();
#pragma unroll
    for (int off = 128; off > 0; off >>= 1) {
        if (t < off) sbf[t] += sbf[t + off];
        __syncthreads();
    }
    if (t == 0) lossArr[cbase + slot] = sbf[0];
}

__global__ void finalize_k(const unsigned* __restrict__ gts, const float* __restrict__ loss,
                           float* __restrict__ out) {
    float tot = 0.f;
    int n = 0;
    for (int c = 0; c < NCLS; c++)
        if (gts[c] > 0u) { tot += loss[c]; n++; }
    out[0] = (n > 0) ? (tot / (float)n) : 0.f;
}

// -------------------------------------------------------------------------
// Host launcher
// -------------------------------------------------------------------------
extern "C" void kernel_launch(void* const* d_in, const int* in_sizes, int n_in,
                              void* d_out, int out_size, void* d_ws, size_t ws_size,
                              hipStream_t stream) {
    (void)in_sizes; (void)n_in; (void)out_size;
    const float* logits = (const float*)d_in[0];
    const int* tgt = (const int*)d_in[1];
    float* out = (float*)d_out;

    auto need = [](int g) -> size_t {
        size_t per = (size_t)2 * PPIX * 4           // key ping-pong
                   + (size_t)2 * 16 * NBLK * 4      // hist + scanned base
                   + (size_t)3 * NBLK * 4;          // blockFg, blockFgBase, blockLoss
        return 4096 + (size_t)g * per;
    };
    int g = NCLS;
    while (g > 1 && need(g) > ws_size) g--;

    char* w = (char*)d_ws;
    size_t off = 0;
    auto alloc = [&](size_t bytes) -> char* {
        char* r = w + off;
        off += (bytes + 255) & ~(size_t)255;
        return r;
    };
    unsigned* gtsArr      = (unsigned*)alloc(NCLS * sizeof(unsigned));
    float*    lossArr     = (float*)   alloc(NCLS * sizeof(float));
    unsigned* bufA        = (unsigned*)alloc((size_t)g * PPIX * sizeof(unsigned));
    unsigned* bufB        = (unsigned*)alloc((size_t)g * PPIX * sizeof(unsigned));
    unsigned* hist        = (unsigned*)alloc((size_t)g * 16 * NBLK * sizeof(unsigned));
    unsigned* baseA       = (unsigned*)alloc((size_t)g * 16 * NBLK * sizeof(unsigned));
    unsigned* blockFg     = (unsigned*)alloc((size_t)g * NBLK * sizeof(unsigned));
    unsigned* blockFgBase = (unsigned*)alloc((size_t)g * NBLK * sizeof(unsigned));
    float*    blockLoss   = (float*)   alloc((size_t)g * NBLK * sizeof(float));

    for (int cbase = 0; cbase < NCLS;) {
        int gg = (NCLS - cbase < g) ? (NCLS - cbase) : g;
        build_keys<<<PPIX / 256, 256, 0, stream>>>(logits, tgt, bufA, cbase, gg);
        unsigned* s = bufA;
        unsigned* d = bufB;
        for (int pass = 0; pass < 8; pass++) {
            radix_hist<<<dim3(NBLK, gg), 256, 0, stream>>>(s, hist, pass);
            scan_hist<<<gg, 256, 0, stream>>>(hist, baseA);
            radix_scatter<<<dim3(NBLK, gg), 256, 0, stream>>>(s, d, baseA, pass);
            unsigned* tmp = s; s = d; d = tmp;
        }
        // 8 passes -> sorted data back in bufA (== s).
        fg_count<<<dim3(NBLK, gg), 256, 0, stream>>>(s, blockFg);
        scan_fg<<<gg, 256, 0, stream>>>(blockFg, blockFgBase, gtsArr, cbase);
        lovasz_block<<<dim3(NBLK, gg), 256, 0, stream>>>(s, blockFgBase, gtsArr, blockLoss, cbase);
        reduce_loss<<<gg, 256, 0, stream>>>(blockLoss, lossArr, cbase);
        cbase += gg;
    }
    finalize_k<<<1, 1, 0, stream>>>(gtsArr, lossArr, out);
}
